// RNN_67130338836668
// MI455X (gfx1250) — matrix operand order repaired and checked by reading, not compiled, not run on hardware
//
#include <hip/hip_runtime.h>
#include <hip/hip_bf16.h>
#include <math.h>

typedef __attribute__((ext_vector_type(16))) __bf16 v16bf;
typedef __attribute__((ext_vector_type(8)))  __bf16 v8bf;
typedef __attribute__((ext_vector_type(8)))  float  v8f;

#define B_   32
#define T_   1024
#define D_   512
#define H_   1024
#define M_   (B_ * T_)          // 32768 rows = (b*T + t)
#define NWG_SCAN 8              // 8 WGs, each owns a 128-col slice of W_hh in LDS
#define PITCH (H_ + 8)          // bf16 elems per LDS row: 2064B pitch -> bank stagger, 16B aligned

// ---------------- elementwise prep kernels ----------------

__global__ void cvt_f32_bf16(__bf16* __restrict__ dst, const float* __restrict__ src, int n) {
  int i = blockIdx.x * blockDim.x + threadIdx.x;
  int stride = gridDim.x * blockDim.x;
  for (; i < n; i += stride) dst[i] = (__bf16)src[i];
}

__global__ void bias_sum(float* __restrict__ dst, const float* __restrict__ a,
                         const float* __restrict__ b, int n) {
  int i = blockIdx.x * blockDim.x + threadIdx.x;
  if (i < n) dst[i] = a[i] + b[i];
}

__global__ void scan_init(__bf16* __restrict__ hbuf, unsigned* __restrict__ bar) {
  int i = blockIdx.x * blockDim.x + threadIdx.x;
  if (i < 2 * B_ * H_) hbuf[i] = (__bf16)0.0f;
  if (i < 64) bar[i] = 0u;
}

// ---- fragment builders (contiguous k order, shared by A and B) ----
// A 16x32 bf16: lane (m = l15), 16 contiguous K elems at K+16*hi
// B 32x16 bf16: lane (n = l15), 16 contiguous K elems at K+16*hi

static __device__ inline v16bf load_a_frag(const __bf16* arow, int k0, int hi) {
  v8bf alo = *(const v8bf*)(arow + k0 + hi * 16);
  v8bf ahi = *(const v8bf*)(arow + k0 + hi * 16 + 8);
  return __builtin_shufflevector(alo, ahi, 0,1,2,3,4,5,6,7,8,9,10,11,12,13,14,15);
}

// ---------------- WMMA GEMM: out[M,N] = A[M,K](bf16) @ W[N,K]^T(bf16) + bias[N] ----------------
// 2x2 register blocking: each wave computes a 32x32 block (4 accumulators),
// so 8 x b128 loads feed 4 WMMAs (2 loads/WMMA instead of 4).

__global__ __launch_bounds__(256) void gemm_pre(
    float* __restrict__ out, const __bf16* __restrict__ A,
    const __bf16* __restrict__ W, const float* __restrict__ bias,
    int M, int N, int K)
{
  const int lane = threadIdx.x & 31;
  const int w    = threadIdx.x >> 5;
  const int hi   = lane >> 4;
  const int l15  = lane & 15;
  const int m0   = blockIdx.y * 64 + (w & 1) * 32;    // wave owns rows [m0, m0+32)
  const int n0   = blockIdx.x * 128 + (w >> 1) * 32;  // wave owns cols [n0, n0+32)

  const __bf16* arow0 = A + (size_t)(m0 + l15) * K;
  const __bf16* arow1 = arow0 + (size_t)16 * K;
  const __bf16* brow0 = W + (size_t)(n0 + l15) * K;
  const __bf16* brow1 = brow0 + (size_t)16 * K;

  v8f acc00 = {}, acc01 = {}, acc10 = {}, acc11 = {};
  for (int k0 = 0; k0 < K; k0 += 32) {
    v16bf a0 = load_a_frag(arow0, k0, hi);
    v16bf a1 = load_a_frag(arow1, k0, hi);
    v16bf b0 = *(const v16bf*)(brow0 + k0 + hi * 16);
    v16bf b1 = *(const v16bf*)(brow1 + k0 + hi * 16);
    acc00 = __builtin_amdgcn_wmma_f32_16x16x32_bf16(false, a0, false, b0, (short)0, acc00, false, false);
    acc01 = __builtin_amdgcn_wmma_f32_16x16x32_bf16(false, a0, false, b1, (short)0, acc01, false, false);
    acc10 = __builtin_amdgcn_wmma_f32_16x16x32_bf16(false, a1, false, b0, (short)0, acc10, false, false);
    acc11 = __builtin_amdgcn_wmma_f32_16x16x32_bf16(false, a1, false, b1, (short)0, acc11, false, false);
  }

  const int na = n0 + l15;
  const int nb = na + 16;
  const float bn_a = bias[na];
  const float bn_b = bias[nb];
#pragma unroll
  for (int r = 0; r < 8; ++r) {
    int m = m0 + r + 8 * hi;                 // C/D layout: M = r + 8*(lane>>4), N = lane&15
    out[(size_t)m * N + na]        = acc00[r] + bn_a;
    out[(size_t)m * N + nb]        = acc01[r] + bn_b;
    out[(size_t)(m + 16) * N + na] = acc10[r] + bn_a;
    out[(size_t)(m + 16) * N + nb] = acc11[r] + bn_b;
  }
}

// ---------------- persistent recurrent scan ----------------

__device__ inline void grid_barrier(unsigned* cnt, unsigned* gen, unsigned nwg) {
  __threadfence();            // make this thread's global stores device-visible
  __syncthreads();
  if (threadIdx.x == 0) {
    unsigned g = __hip_atomic_load(gen, __ATOMIC_ACQUIRE, __HIP_MEMORY_SCOPE_AGENT);
    unsigned a = __hip_atomic_fetch_add(cnt, 1u, __ATOMIC_ACQ_REL, __HIP_MEMORY_SCOPE_AGENT);
    if (a == nwg - 1u) {
      __hip_atomic_store(cnt, 0u, __ATOMIC_RELAXED, __HIP_MEMORY_SCOPE_AGENT);
      __hip_atomic_fetch_add(gen, 1u, __ATOMIC_ACQ_REL, __HIP_MEMORY_SCOPE_AGENT);
    } else {
      while (__hip_atomic_load(gen, __ATOMIC_ACQUIRE, __HIP_MEMORY_SCOPE_AGENT) == g)
        __builtin_amdgcn_s_sleep(2);
    }
  }
  __syncthreads();
}

// 8 workgroups x 256 threads. Each WG stages a 128-column slice of W_hh into
// ~258 KB of the WGP's 320 KB LDS. Each wave owns one 16-col subtile and both
// 16-row batch blocks (M-pair), so each LDS B fragment feeds two WMMAs.

__global__ __launch_bounds__(256) void rnn_scan(
    const float*  __restrict__ pre,     // [M_, H_], row = b*T + t
    const __bf16* __restrict__ Whh,     // [H_, H_] bf16 row-major (n, k)
    float*        __restrict__ y_f32,   // layer1 -> d_out, else nullptr
    __bf16*       __restrict__ y_bf16,  // layer0 -> y0b,   else nullptr
    float*        __restrict__ hid_out, // final hidden slot for this layer [B_, H_]
    __bf16*       __restrict__ hbuf,    // double buffer: 2 * [B_, H_] bf16
    unsigned*     __restrict__ bar)
{
  extern __shared__ __bf16 lds[];       // 128 rows x PITCH bf16
  const int lane = threadIdx.x & 31;
  const int w    = threadIdx.x >> 5;
  const int hi   = lane >> 4;
  const int l15  = lane & 15;
  const int nl0  = w * 16;              // local column block inside WG slice
  const int nwg0 = blockIdx.x * 128;    // this WG's column slice of W_hh
  const int n    = nwg0 + nl0 + l15;

  // Stage W_hh[slice rows] into LDS, vectorized 16B copies.
  for (int i = threadIdx.x; i < 128 * (H_ / 8); i += blockDim.x) {
    int row = i >> 7;                   // 128 8-elem chunks per 1024-wide row
    int c8  = (i & 127) * 8;
    *(v8bf*)&lds[row * PITCH + c8] = *(const v8bf*)&Whh[(size_t)(nwg0 + row) * H_ + c8];
  }
  __syncthreads();

  const __bf16* brow = &lds[(nl0 + l15) * PITCH];

  for (int t = 0; t < T_; ++t) {
    const __bf16* hcur = hbuf + (size_t)(t & 1) * (B_ * H_);
    __bf16*       hnxt = hbuf + (size_t)((t + 1) & 1) * (B_ * H_);
    const __bf16* arow0 = hcur + (size_t)l15 * H_;          // batch rows 0..15
    const __bf16* arow1 = hcur + (size_t)(16 + l15) * H_;   // batch rows 16..31

    v8f acc0 = {}, acc1 = {};
#pragma unroll 4
    for (int k0 = 0; k0 < H_; k0 += 32) {
      v16bf a0  = load_a_frag(arow0, k0, hi);
      v16bf a1  = load_a_frag(arow1, k0, hi);
      v16bf bfg = *(const v16bf*)(brow + k0 + hi * 16);     // ds_load_b128 x2, reused twice
      acc0 = __builtin_amdgcn_wmma_f32_16x16x32_bf16(false, a0, false, bfg, (short)0, acc0, false, false);
      acc1 = __builtin_amdgcn_wmma_f32_16x16x32_bf16(false, a1, false, bfg, (short)0, acc1, false, false);
    }

#pragma unroll
    for (int r = 0; r < 8; ++r) {
      int m = r + 8 * hi;                            // batch index (block 0)
      {
        size_t row = (size_t)m * T_ + t;
        float v = tanhf(acc0[r] + pre[row * H_ + n]);
        hnxt[(size_t)m * H_ + n] = (__bf16)v;
        if (y_f32)  y_f32[row * H_ + n] = v;
        if (y_bf16) y_bf16[row * H_ + n] = (__bf16)v;
        if (t == T_ - 1) hid_out[(size_t)m * H_ + n] = v;
      }
      {
        int m2 = m + 16;                             // batch index (block 1)
        size_t row = (size_t)m2 * T_ + t;
        float v = tanhf(acc1[r] + pre[row * H_ + n]);
        hnxt[(size_t)m2 * H_ + n] = (__bf16)v;
        if (y_f32)  y_f32[row * H_ + n] = v;
        if (y_bf16) y_bf16[row * H_ + n] = (__bf16)v;
        if (t == T_ - 1) hid_out[(size_t)m2 * H_ + n] = v;
      }
    }
    grid_barrier(bar, bar + 32, NWG_SCAN);           // 8 WGs, trivially co-resident
  }
}

// ---------------- launcher ----------------

extern "C" void kernel_launch(void* const* d_in, const int* in_sizes, int n_in,
                              void* d_out, int out_size, void* d_ws, size_t ws_size,
                              hipStream_t stream)
{
  (void)in_sizes; (void)n_in; (void)out_size; (void)ws_size;
  const float* x    = (const float*)d_in[0];
  const float* Wih0 = (const float*)d_in[1];
  const float* Whh0 = (const float*)d_in[2];
  const float* bih0 = (const float*)d_in[3];
  const float* bhh0 = (const float*)d_in[4];
  const float* Wih1 = (const float*)d_in[5];
  const float* Whh1 = (const float*)d_in[6];
  const float* bih1 = (const float*)d_in[7];
  const float* bhh1 = (const float*)d_in[8];

  char* ws = (char*)d_ws;
  size_t off = 0;
  auto take = [&](size_t bytes) -> char* {
    char* p = ws + off;
    off = (off + bytes + 255) & ~(size_t)255;
    return p;
  };

  __bf16* xb    = (__bf16*)take((size_t)M_ * D_ * 2);
  __bf16* wih0b = (__bf16*)take((size_t)H_ * D_ * 2);
  __bf16* whh0b = (__bf16*)take((size_t)H_ * H_ * 2);
  __bf16* wih1b = (__bf16*)take((size_t)H_ * H_ * 2);
  __bf16* whh1b = (__bf16*)take((size_t)H_ * H_ * 2);
  float*  bias0 = (float*)take((size_t)H_ * 4);
  float*  bias1 = (float*)take((size_t)H_ * 4);
  float*  pre   = (float*)take((size_t)M_ * H_ * 4);
  __bf16* y0b   = (__bf16*)take((size_t)M_ * H_ * 2);
  __bf16* hbuf  = (__bf16*)take((size_t)2 * B_ * H_ * 2);
  unsigned* bar = (unsigned*)take(256);

  float* y1  = (float*)d_out;                    // [B, T, H]
  float* hid = y1 + (size_t)M_ * H_;             // [2, B, H]

  const int thr = 256;
  cvt_f32_bf16<<<2048, thr, 0, stream>>>(xb,    x,    M_ * D_);
  cvt_f32_bf16<<<512,  thr, 0, stream>>>(wih0b, Wih0, H_ * D_);
  cvt_f32_bf16<<<1024, thr, 0, stream>>>(whh0b, Whh0, H_ * H_);
  cvt_f32_bf16<<<1024, thr, 0, stream>>>(whh1b, Whh1, H_ * H_);
  cvt_f32_bf16<<<1024, thr, 0, stream>>>(wih1b, Wih1, H_ * H_);
  bias_sum<<<(H_ + thr - 1) / thr, thr, 0, stream>>>(bias0, bih0, bhh0, H_);
  bias_sum<<<(H_ + thr - 1) / thr, thr, 0, stream>>>(bias1, bih1, bhh1, H_);

  const size_t ldsB = (size_t)128 * PITCH * sizeof(__bf16);   // ~258 KB of 320 KB WGP LDS

  // ----- layer 0 -----
  scan_init<<<(2 * B_ * H_ + thr - 1) / thr, thr, 0, stream>>>(hbuf, bar);
  gemm_pre<<<dim3(H_ / 128, M_ / 64), thr, 0, stream>>>(pre, xb, wih0b, bias0, M_, H_, D_);
  rnn_scan<<<NWG_SCAN, thr, ldsB, stream>>>(pre, whh0b, nullptr, y0b, hid, hbuf, bar);

  // ----- layer 1 -----
  scan_init<<<(2 * B_ * H_ + thr - 1) / thr, thr, 0, stream>>>(hbuf, bar);
  gemm_pre<<<dim3(H_ / 128, M_ / 64), thr, 0, stream>>>(pre, y0b, wih1b, bias1, M_, H_, H_);
  rnn_scan<<<NWG_SCAN, thr, ldsB, stream>>>(pre, whh1b, y1, nullptr, hid + (size_t)B_ * H_, hbuf, bar);
}
